// VLSTMSDEEncoder_1211180778443
// MI455X (gfx1250) — matrix operand order repaired and checked by reading, not compiled, not run on hardware
//
#include <hip/hip_runtime.h>

#define BB   16
#define TT   256
#define NN   32
#define HH   128
#define LL   2
#define DDIM 257     // 2H+1
#define DP   288     // D padded to multiple of 32
#define FH   512     // 4H
#define TH3  384     // 3H
#define RB   16      // rows per block (one WMMA M tile)
#define THREADS 256
#define KT_D 9       // 288/32 K-tiles for D-dim matmuls
#define KT_H 4       // 128/32 K-tiles
#define NT_FH 32
#define NT_H  8
#define NT_3H 24

typedef __bf16 bf16_t;
typedef __bf16 v16bf __attribute__((ext_vector_type(16)));
typedef __bf16 v8bf  __attribute__((ext_vector_type(8)));
typedef float  v8f   __attribute__((ext_vector_type(8)));

// ---------------------------------------------------------------------------
// Weight pre-pack: f32 [L][K][Ntot] row-major  ->  bf16 WMMA B-fragment blocks
// block (l,kt,nt): 512 elems; elem = ln*16 + j  <->  B[kt*32 + (ln<16?0:8) + (j<8?j:j+8)][nt*16 + ln%16]
// so the main kernel loads a whole v16bf B-fragment as one contiguous 32B read.
// ---------------------------------------------------------------------------
__global__ void pack_weights(const float* __restrict__ src, bf16_t* __restrict__ dst,
                             int K, int KT, int Ntot, int NT, int total)
{
    for (int idx = blockIdx.x * blockDim.x + threadIdx.x; idx < total;
         idx += gridDim.x * blockDim.x) {
        int e   = idx & 511;
        int blk = idx >> 9;
        int nt  = blk % NT; blk /= NT;
        int kt  = blk % KT;
        int l   = blk / KT;
        int ln  = e >> 4;
        int j   = e & 15;
        int k   = kt * 32 + ((ln < 16) ? 0 : 8) + ((j < 8) ? j : j + 8);
        int n   = nt * 16 + (ln & 15);
        float v = 0.0f;
        if (k < K && n < Ntot) v = src[((size_t)l * K + k) * Ntot + n];
        dst[idx] = (bf16_t)v;
    }
}

// A-fragment loader: LDS row-major bf16 matrix [16][rowStride], 16-bit A layout:
// lane<16 : M=lane,   K = ktOff + {0..7, 16..23}
// lane>=16: M=lane-16,K = ktOff + {8..15, 24..31}
// => two contiguous 16B LDS loads per lane (ds_load_b128 x2)
__device__ __forceinline__ v16bf load_a_frag(const bf16_t* __restrict__ base,
                                             int rowStride, int ln, int ktOff)
{
    int m  = ln & 15;
    int kb = ktOff + ((ln < 16) ? 0 : 8);
    const bf16_t* p = base + m * rowStride + kb;
    v8bf lo = *(const v8bf*)p;
    v8bf hi = *(const v8bf*)(p + 16);
    v16bf a;
#pragma unroll
    for (int j = 0; j < 8; ++j) { a[j] = lo[j]; a[j + 8] = hi[j]; }
    return a;
}

__device__ __forceinline__ v16bf load_b_frag(const bf16_t* __restrict__ blockBase, int ln)
{
    return *(const v16bf*)(blockBase + ln * 16);
}

__device__ __forceinline__ v8f wmma_bf16(v16bf a, v16bf b, v8f c)
{
    return __builtin_amdgcn_wmma_f32_16x16x32_bf16(false, a, false, b, (short)0, c,
                                                   false, false);
}

// Fast transcendental helpers: single v_exp_f32 + v_rcp_f32, inputs clamped so
// exp never overflows (|x|<=16 covers full fp32 sigmoid/tanh saturation).
__device__ __forceinline__ float fast_sigmoid(float x)
{
    x = fminf(fmaxf(x, -16.0f), 16.0f);
    return __builtin_amdgcn_rcpf(1.0f + __expf(-x));
}
__device__ __forceinline__ float fast_tanh(float x)
{
    x = fminf(fmaxf(x, -8.0f), 8.0f);
    float e = __expf(2.0f * x);
    return (e - 1.0f) * __builtin_amdgcn_rcpf(e + 1.0f);
}

// ---------------------------------------------------------------------------
// Persistent scan kernel: one workgroup owns 16 (b,n) rows for all T steps.
// Two persistent bf16 gate_in buffers (one per layer); producers write their
// results directly into final gate_in slots -> no assembly/copy phase.
//   sgin[l] = [ layer_in (0:128) | h[l] (128:256) | dt (256) | zero-pad (257:288) ]
//   h_new region doubles as the Wo A-matrix (read at column offset +128).
// Weights are STREAMED from L2 every step (they're permanently L2-resident);
// an explicit per-iteration memory barrier stops LICM from hoisting the
// fragment loads out of the scan loop (which previously caused scratch spills
// feeding every WMMA from swizzled private memory).
// ---------------------------------------------------------------------------
__global__ __launch_bounds__(THREADS)
__attribute__((amdgpu_waves_per_eu(2)))
void vlstm_scan(const float* __restrict__ x, const float* __restrict__ dt,
                const float* __restrict__ mask,
                const float* __restrict__ node_w, const float* __restrict__ node_b,
                const float* __restrict__ bg, const float* __restrict__ b1,
                const float* __restrict__ b2, const float* __restrict__ bo,
                const float* __restrict__ noise,
                const bf16_t* __restrict__ WgP, const bf16_t* __restrict__ W1P,
                const bf16_t* __restrict__ W2P, const bf16_t* __restrict__ WoP,
                float* __restrict__ out)
{
    extern __shared__ char smem[];
    float*  sh    = (float*)smem;                // [LL][RB][HH] hidden state (f32)
    float*  sc    = sh + LL * RB * HH;           // [LL][RB][HH] cell state   (f32)
    float*  sdt   = sc + LL * RB * HH;           // [RB] per-row dt at step t
    bf16_t* sgin0 = (bf16_t*)(sdt + RB);         // [RB][DP] gate_in layer 0
    bf16_t* sgin1 = sgin0 + RB * DP;             // [RB][DP] gate_in layer 1
    bf16_t* sru   = sgin1 + RB * DP;             // [RB][HH] relu(W1) output

    const int tid  = threadIdx.x;
    const int wave = tid >> 5;
    const int ln   = tid & 31;
    const int r0   = blockIdx.x * RB;
    const int colw = wave * 16 + (ln & 15);      // this lane's output column

    // per-thread mapping for the xin phase (16 threads per row, 8 cols each)
    const int em   = tid >> 4;
    const int ej0  = (tid & 15) * 8;
    const int erow = r0 + em;
    const int eb   = erow >> 5, en = erow & 31;

    // --- init: zero states and both gate_in buffers (incl. h and pad columns) ---
    for (int i = tid; i < 2 * LL * RB * HH; i += THREADS) sh[i] = 0.0f; // sh+sc
    for (int i = tid; i < 2 * RB * DP; i += THREADS) sgin0[i] = (bf16_t)0.0f;

    // --- hoist loop invariants into registers ---
    float Bgf[LL], Bgi[LL], Bgo[LL], Bgc[LL], B2a[LL], B2o[LL], B2p[LL], B1v[LL], Bov[LL];
#pragma unroll
    for (int l = 0; l < LL; ++l) {
        Bgf[l] = bg[l * FH + colw];
        Bgi[l] = bg[l * FH + HH + colw];
        Bgo[l] = bg[l * FH + 2 * HH + colw];
        Bgc[l] = bg[l * FH + 3 * HH + colw];
        B2a[l] = b2[l * TH3 + colw];
        B2o[l] = b2[l * TH3 + HH + colw];
        B2p[l] = b2[l * TH3 + 2 * HH + colw];
        B1v[l] = b1[l * HH + colw];
        Bov[l] = bo[l * HH + colw];
    }
    float nwv[8], nbv[8];
#pragma unroll
    for (int q = 0; q < 8; ++q) {
        nwv[q] = node_w[en * HH + ej0 + q];
        nbv[q] = node_b[en * HH + ej0 + q];
    }
    // loop-invariant flat index bases (32-bit: out has 2^24 elements)
    const int xmBase = eb * TT * NN + en;                 // + t*NN per step
    int obase[8];                                         // out offset sans t-term
#pragma unroll
    for (int r = 0; r < 8; ++r) {
        int mr = (ln < 16) ? r : r + 8;
        int rr = r0 + mr;
        obase[r] = ((rr >> 5) * TT * NN + (rr & 31)) * HH + colw;  // + t*NN*HH
    }
    const int dtBase = (tid < RB) ? (((r0 + tid) >> 5) * TT * NN + ((r0 + tid) & 31)) : 0;

    __syncthreads();

    for (int t = 0; t < TT; ++t) {
        // Compiler fence: forbid hoisting the (loop-invariant) weight-fragment
        // loads out of the scan loop. Streaming them from L2 each iteration is
        // cheap; hoisting them spills ~500 VGPRs of fragments to scratch.
        asm volatile("" ::: "memory");

        // --- xin = (x*node_w + node_b)*mask -> sgin0[:,0:128]; dt -> col 256 + sdt ---
        {
            float xv = x[xmBase + t * NN];
            float mv = mask[xmBase + t * NN];
#pragma unroll
            for (int q = 0; q < 8; ++q)
                sgin0[em * DP + ej0 + q] = (bf16_t)((xv * nwv[q] + nbv[q]) * mv);
            if (tid < RB) {
                float d = dt[dtBase + t * NN];
                sdt[tid] = d;
                sgin0[tid * DP + 2 * HH] = (bf16_t)d;
                sgin1[tid * DP + 2 * HH] = (bf16_t)d;
            }
        }

#pragma unroll
        for (int l = 0; l < LL; ++l) {
            bf16_t* gin = (l == 0) ? sgin0 : sgin1;
            __syncthreads();  // gate_in[l] fully assembled

            // --- Wg (gate-aligned tiles w,w+8,w+16,w+24) and W1 (tile w), shared A ---
            v8f af = {}, ai = {}, ao = {}, ac = {}, a1 = {};
#pragma unroll
            for (int kt = 0; kt < KT_D; ++kt) {
                v16bf a = load_a_frag(gin, DP, ln, kt * 32);
                const bf16_t* gb = WgP + (((size_t)l * KT_D + kt) * NT_FH) * 512;
                af = wmma_bf16(a, load_b_frag(gb + (size_t)(wave)      * 512, ln), af);
                ai = wmma_bf16(a, load_b_frag(gb + (size_t)(wave + 8)  * 512, ln), ai);
                ao = wmma_bf16(a, load_b_frag(gb + (size_t)(wave + 16) * 512, ln), ao);
                ac = wmma_bf16(a, load_b_frag(gb + (size_t)(wave + 24) * 512, ln), ac);
                const bf16_t* wb = W1P + (((size_t)l * KT_D + kt) * NT_H + wave) * 512;
                a1 = wmma_bf16(a, load_b_frag(wb, ln), a1);
            }
            // relu(W1 + b1) -> sru (bf16)
#pragma unroll
            for (int r = 0; r < 8; ++r) {
                int mr = (ln < 16) ? r : r + 8;
                sru[mr * HH + colw] = (bf16_t)fmaxf(a1[r] + B1v[l], 0.0f);
            }
            __syncthreads();  // sru ready (also fences gate_in reads before h rewrite)

            // --- W2: tiles w, w+8, w+16 (A, omega, phi aligned per lane) ---
            v8f mA = {}, mO = {}, mP = {};
#pragma unroll
            for (int kt = 0; kt < KT_H; ++kt) {
                v16bf a = load_a_frag(sru, HH, ln, kt * 32);
                const bf16_t* wb = W2P + (((size_t)l * KT_H + kt) * NT_3H) * 512;
                mA = wmma_bf16(a, load_b_frag(wb + (size_t)(wave)      * 512, ln), mA);
                mO = wmma_bf16(a, load_b_frag(wb + (size_t)(wave + 8)  * 512, ln), mO);
                mP = wmma_bf16(a, load_b_frag(wb + (size_t)(wave + 16) * 512, ln), mP);
            }

            // --- register-resident SDE elementwise update ---
            {
                // noise flat index: ((t*L + l)*512 + row)*128 + colw, row = r0+mr
                size_t nbase = ((size_t)(t * LL + l) * (BB * NN) + r0 +
                                ((ln < 16) ? 0 : 8)) * HH + colw;
#pragma unroll
                for (int r = 0; r < 8; ++r) {
                    int mr = (ln < 16) ? r : r + 8;
                    float dtc = sdt[mr];
                    float dtm = fmaxf(dtc, 1e-4f);
                    float rdt = __builtin_amdgcn_rcpf(dtm);   // v_rcp_f32
                    float fg = fast_sigmoid(af[r] + Bgf[l]);
                    float ig = fast_sigmoid(ai[r] + Bgi[l]);
                    float og = fast_sigmoid(ao[r] + Bgo[l]);
                    float ct = fast_tanh(ac[r] + Bgc[l]);
                    float Am = mA[r] + B2a[l];
                    float om = mO[r] + B2o[l];
                    float ph = mP[r] + B2p[l];
                    float z  = Am + Am * __sinf(om * dtc + ph);
                    float brown = noise[nbase + (size_t)r * HH] * (0.05f * sqrtf(dtm));
                    int hi = (l * RB + mr) * HH + colw;
                    float cold = sc[hi], hold = sh[hi];
                    float cnew = cold +
                        ((-(1.0f - fg) * rdt) * cold + (ig * rdt) * ct + z) * dtc + brown;
                    float hnew = hold + (og * fast_tanh(cnew) - 0.1f * hold) * dtc;
                    sc[hi] = cnew;
                    sh[hi] = hnew;
                    gin[mr * DP + HH + colw] = (bf16_t)hnew;   // next step's h slot = Wo A
                }
            }
            __syncthreads();  // h_new (bf16) ready in gate_in[l][:,128:256]

            // --- Wo: layer output; A-matrix is the h_new region of gate_in[l] ---
            v8f aw = {};
#pragma unroll
            for (int kt = 0; kt < KT_H; ++kt) {
                v16bf a = load_a_frag(gin + HH, DP, ln, kt * 32);
                const bf16_t* wb = WoP + (((size_t)l * KT_H + kt) * NT_H + wave) * 512;
                aw = wmma_bf16(a, load_b_frag(wb, ln), aw);
            }
            if (l == 0) {
                // becomes layer 1's input: write bf16 straight into sgin1[:,0:128]
#pragma unroll
                for (int r = 0; r < 8; ++r) {
                    int mr = (ln < 16) ? r : r + 8;
                    sgin1[mr * DP + colw] = (bf16_t)(aw[r] + Bov[0]);
                }
            } else {
                // final output for step t: flat offset = obase[r] + t*N*H
                const int toff = t * (NN * HH);
#pragma unroll
                for (int r = 0; r < 8; ++r)
                    out[obase[r] + toff] = aw[r] + Bov[1];
            }
        }
        // no end-of-step barrier needed: next xin touches only sgin0[:,0:128],
        // dt columns and sdt, none of which are read by the trailing Wo phase.
    }
}

extern "C" void kernel_launch(void* const* d_in, const int* in_sizes, int n_in,
                              void* d_out, int out_size, void* d_ws, size_t ws_size,
                              hipStream_t stream)
{
    const float* x      = (const float*)d_in[0];
    const float* dt     = (const float*)d_in[1];
    const float* mask   = (const float*)d_in[2];
    const float* node_w = (const float*)d_in[3];
    const float* node_b = (const float*)d_in[4];
    const float* Wg     = (const float*)d_in[5];
    const float* bg     = (const float*)d_in[6];
    const float* W1     = (const float*)d_in[7];
    const float* b1     = (const float*)d_in[8];
    const float* W2     = (const float*)d_in[9];
    const float* b2     = (const float*)d_in[10];
    const float* Wo     = (const float*)d_in[11];
    const float* bo     = (const float*)d_in[12];
    const float* noise  = (const float*)d_in[13];
    float* out = (float*)d_out;

    bf16_t* WgP = (bf16_t*)d_ws;
    bf16_t* W1P = WgP + (size_t)LL * KT_D * NT_FH * 512;
    bf16_t* W2P = W1P + (size_t)LL * KT_D * NT_H  * 512;
    bf16_t* WoP = W2P + (size_t)LL * KT_H * NT_3H * 512;

    pack_weights<<<128, 256, 0, stream>>>(Wg, WgP, DDIM, KT_D, FH,  NT_FH, LL * KT_D * NT_FH * 512);
    pack_weights<<<128, 256, 0, stream>>>(W1, W1P, DDIM, KT_D, HH,  NT_H,  LL * KT_D * NT_H  * 512);
    pack_weights<<<128, 256, 0, stream>>>(W2, W2P, HH,   KT_H, TH3, NT_3H, LL * KT_H * NT_3H * 512);
    pack_weights<<<128, 256, 0, stream>>>(Wo, WoP, HH,   KT_H, HH,  NT_H,  LL * KT_H * NT_H  * 512);

    // LDS: h + c + sdt (f32) ; gin0 + gin1 + ru (bf16)
    size_t smem = (size_t)(LL * RB * HH * 2 + RB) * sizeof(float) +
                  (size_t)(2 * RB * DP + RB * HH) * sizeof(bf16_t);

    vlstm_scan<<<(BB * NN) / RB, THREADS, smem, stream>>>(
        x, dt, mask, node_w, node_b, bg, b1, b2, bo, noise, WgP, W1P, W2P, WoP, out);
}
